// LearnedEdgeConstructor_11132555231562
// MI455X (gfx1250) — matrix operand order repaired and checked by reading, not compiled, not run on hardware
//
#include <hip/hip_runtime.h>
#include <hip/hip_bf16.h>

// ---------------- problem constants (from reference) ----------------
#define NODES   8192
#define FDIM    64
#define HDIM    128
#define RREG    3
#define DEG     32
#define KTOP    32
#define EDGES   (NODES*DEG)        // 262144
#define DRAW    131                // 2F+3
#define DPAD    160                // padded to 5 * 32 (bf16 WMMA K)
#define TEMP_E  0.3f
#define NEGSL   0.2f
#define EPS_LN  1e-5f
#define W_THR   1e-6f

// ---------------- workspace layout (float dwords) ----------------
#define WS_AMP     1
#define WS_RP      4                       // 3 regime probs
#define WS_GS      8                       // 64 global mean
#define WS_WS1     128                     // 5*8*256 = 10240 dwords (bf16 B frags)
#define WS_WP      (WS_WS1 + 10240)
#define WS_WG1     (WS_WP  + 10240)        // 3*4*4*256 = 12288 dwords
#define WS_SCORES  (WS_WG1 + 12288)        // E floats
#define WS_CONTN   WS_SCORES               // per-node contagion (consumed before scores written)

typedef __attribute__((ext_vector_type(16))) __bf16 v16bf;
typedef __attribute__((ext_vector_type(8)))  float  v8f;
typedef union { v16bf v; uint4 u[2]; } frag_t;

// ---------------- column means of x -> gs[64] ----------------
__global__ void colmean_k(const float* __restrict__ x, float* __restrict__ wsf) {
  __shared__ float red[256];
  int f = blockIdx.x, t = threadIdx.x;
  float s = 0.f;
  for (int n = t; n < NODES; n += 256) s += x[n*FDIM + f];
  red[t] = s; __syncthreads();
  for (int o = 128; o > 0; o >>= 1) { if (t < o) red[t] += red[t+o]; __syncthreads(); }
  if (t == 0) wsf[WS_GS + f] = red[0] / (float)NODES;
}

// ---------------- per-node contagion sigmoid (one wave / node) ----------------
__global__ void contagion_k(const float* __restrict__ x,
                            const float* __restrict__ Wc1, const float* __restrict__ bc1,
                            const float* __restrict__ Wc2, const float* __restrict__ bc2,
                            float* __restrict__ wsf) {
  int gid = blockIdx.x*blockDim.x + threadIdx.x;
  int n = gid >> 5, lane = threadIdx.x & 31;
  if (n >= NODES) return;
  int hd0 = lane, hd1 = lane+32, hd2 = lane+64, hd3 = lane+96;
  float h0 = bc1[hd0], h1 = bc1[hd1], h2 = bc1[hd2], h3 = bc1[hd3];
  for (int f = 0; f < FDIM; f++) {
    float xv = x[n*FDIM + f];
    h0 = fmaf(xv, Wc1[f*HDIM+hd0], h0);
    h1 = fmaf(xv, Wc1[f*HDIM+hd1], h1);
    h2 = fmaf(xv, Wc1[f*HDIM+hd2], h2);
    h3 = fmaf(xv, Wc1[f*HDIM+hd3], h3);
  }
  h0 = h0 > 0.f ? h0 : 0.f; h1 = h1 > 0.f ? h1 : 0.f;
  h2 = h2 > 0.f ? h2 : 0.f; h3 = h3 > 0.f ? h3 : 0.f;
  float a = h0*Wc2[hd0] + h1*Wc2[hd1] + h2*Wc2[hd2] + h3*Wc2[hd3];
  #pragma unroll
  for (int m = 1; m < 32; m <<= 1) a += __shfl_xor(a, m);
  if (lane == 0) wsf[WS_CONTN + n] = 1.f / (1.f + __expf(-(a + bc2[0])));
}

// ---------------- regime softmax + amp (single block, deterministic) ----------------
__global__ void scalars_k(const float* __restrict__ Wr1, const float* __restrict__ br1,
                          const float* __restrict__ Wr2, const float* __restrict__ br2,
                          float* __restrict__ wsf) {
  __shared__ float red[128];
  __shared__ float hh[128];
  int t = threadIdx.x;
  float cs = 0.f;
  for (int n = t; n < NODES; n += 128) cs += wsf[WS_CONTN + n];
  red[t] = cs; __syncthreads();
  for (int o = 64; o > 0; o >>= 1) { if (t < o) red[t] += red[t+o]; __syncthreads(); }
  float cont = red[0] / (float)NODES;
  float h = br1[t];
  for (int f = 0; f < FDIM; f++) h = fmaf(wsf[WS_GS + f], Wr1[f*HDIM + t], h);
  hh[t] = h > 0.f ? h : 0.f;
  __syncthreads();
  if (t == 0) {
    float o[3];
    for (int r = 0; r < 3; r++) {
      float s = br2[r];
      for (int k = 0; k < HDIM; k++) s = fmaf(hh[k], Wr2[k*RREG + r], s);
      o[r] = s;
    }
    float m = fmaxf(o[0], fmaxf(o[1], o[2]));
    float e0 = __expf(o[0]-m), e1 = __expf(o[1]-m), e2 = __expf(o[2]-m);
    float inv = 1.f / (e0 + e1 + e2);
    wsf[WS_RP+0] = e0*inv; wsf[WS_RP+1] = e1*inv; wsf[WS_RP+2] = e2*inv;
    wsf[WS_AMP] = 1.f + 0.5f * cont;
  }
}

// ---------------- pack weights -> bf16 B fragments (lane-major dwords) ----------------
// B 32x16 bf16 packing used here (contiguous k, same order as A): VGPR j, lanes 0-15: K = 2j,2j+1 ; lanes 16-31: K = 16+2j,16+2j+1 ; N = lane%16.
__global__ void pack_k(const float* __restrict__ Ws1, const float* __restrict__ Wp,
                       const float* __restrict__ Wg1, float* __restrict__ wsf) {
  unsigned* wsu = (unsigned*)wsf;
  int i = blockIdx.x*256 + threadIdx.x;          // 0..32767
  float lo, hi; unsigned dst;
  if (i < 20480) {
    int ii = (i < 10240) ? i : i - 10240;
    const float* W = (i < 10240) ? Ws1 : Wp;
    int f = ii >> 8, r = ii & 255, lane = r >> 3, j = r & 7;
    int kt = f >> 3, nt = f & 7;
    int k0  = 32*kt + 16*(lane>>4) + 2*j;
    int col = 16*nt + (lane & 15);
    lo = (k0     < DRAW) ? W[k0*HDIM + col]     : 0.f;
    hi = (k0 + 1 < DRAW) ? W[(k0+1)*HDIM + col] : 0.f;
    dst = ((i < 10240) ? WS_WS1 : WS_WP) + ii;
  } else {
    int ii = i - 20480;
    int f = ii >> 8, r = ii & 255, lane = r >> 3, j = r & 7;
    int rr = f >> 4, kt = (f >> 2) & 3, nt = f & 3;
    int k0  = 32*kt + 16*(lane>>4) + 2*j;
    int col = 16*nt + (lane & 15);
    lo = Wg1[rr*HDIM*64 + k0*64 + col];
    hi = Wg1[rr*HDIM*64 + (k0+1)*64 + col];
    dst = WS_WG1 + ii;
  }
  union { __bf16 h[2]; unsigned u; } pk;
  pk.h[0] = (__bf16)lo; pk.h[1] = (__bf16)hi;
  wsu[dst] = pk.u;
}

// ---------------- main edge kernel: WMMA GEMMs + LN + gates ----------------
__global__ void __launch_bounds__(256) edge_k(
    const float* __restrict__ x, const int* __restrict__ eidx, const float* __restrict__ ea,
    const float* __restrict__ bs1, const float* __restrict__ ln_g, const float* __restrict__ ln_b,
    const float* __restrict__ Ws2, const float* __restrict__ bs2, const float* __restrict__ bp,
    const float* __restrict__ bg1, const float* __restrict__ Wg2, const float* __restrict__ bg2,
    float* __restrict__ wsf)
{
  __shared__ __align__(16) __bf16 ein[8][16][DPAD];   // 40 KB: per-wave edge_input (bf16)
  __shared__ __align__(16) __bf16 ehb[8][16][HDIM];   // 32 KB: per-wave edge_hidden (bf16)
  int w = threadIdx.x >> 5, lane = threadIdx.x & 31;
  int tile = blockIdx.x*8 + w;
  int e0 = tile * 16;
  int half = lane >> 4, l15 = lane & 15;

  // stage edge_input = [x[src] | x[tgt] | edge_attr | 0-pad] as bf16
  for (int r = 0; r < 16; r++) {
    int e = e0 + r;
    int sr = eidx[e], tr = eidx[EDGES + e];
    for (int c = lane; c < DPAD; c += 32) {
      float v;
      if      (c < FDIM)   v = x[sr*FDIM + c];
      else if (c < 2*FDIM) v = x[tr*FDIM + (c - FDIM)];
      else if (c < DRAW)   v = ea[e*3 + (c - 2*FDIM)];
      else                 v = 0.f;
      ein[w][r][c] = (__bf16)v;
    }
  }
  __builtin_amdgcn_wave_barrier();

  // A fragments: 16x32 bf16 per k-tile (lane l15 = row; half selects K sub-range)
  frag_t A[5];
  const __bf16* rp = &ein[w][l15][0];
  #pragma unroll
  for (int kt = 0; kt < 5; kt++) {
    const uint4* q = (const uint4*)(rp + 32*kt + 16*half);
    A[kt].u[0] = q[0]; A[kt].u[1] = q[1];
  }

  // ---- GEMM1: h = ein @ Ws1  (16x160 @ 160x128) ----
  v8f acc[8];
  #pragma unroll
  for (int nt = 0; nt < 8; nt++) acc[nt] = (v8f){0,0,0,0,0,0,0,0};
  const uint4* b1 = (const uint4*)(wsf + WS_WS1);
  #pragma unroll
  for (int kt = 0; kt < 5; kt++) {
    #pragma unroll
    for (int nt = 0; nt < 8; nt++) {
      frag_t B; const uint4* q = b1 + ((kt*8 + nt)*64 + lane*2);
      B.u[0] = q[0]; B.u[1] = q[1];
      acc[nt] = __builtin_amdgcn_wmma_f32_16x16x32_bf16(false, A[kt].v, false, B.v,
                                                        (short)0, acc[nt], false, false);
    }
  }
  // bias + LayerNorm stats (row = v + 8*half, col = 16*nt + l15)
  float ps[8], pq[8];
  #pragma unroll
  for (int v = 0; v < 8; v++) { ps[v] = 0.f; pq[v] = 0.f; }
  #pragma unroll
  for (int nt = 0; nt < 8; nt++) {
    float b = bs1[16*nt + l15];
    #pragma unroll
    for (int v = 0; v < 8; v++) {
      float h = acc[nt][v] + b; acc[nt][v] = h;
      ps[v] += h; pq[v] += h*h;
    }
  }
  #pragma unroll
  for (int v = 0; v < 8; v++) {
    #pragma unroll
    for (int m = 1; m < 16; m <<= 1) { ps[v] += __shfl_xor(ps[v], m); pq[v] += __shfl_xor(pq[v], m); }
  }
  float mean[8], rs[8];
  #pragma unroll
  for (int v = 0; v < 8; v++) {
    mean[v] = ps[v] * (1.f/HDIM);
    float var = pq[v] * (1.f/HDIM) - mean[v]*mean[v];
    rs[v] = rsqrtf(var + EPS_LN);
  }
  // LN + LeakyReLU + dot(Ws2)
  float raw[8];
  #pragma unroll
  for (int v = 0; v < 8; v++) raw[v] = 0.f;
  #pragma unroll
  for (int nt = 0; nt < 8; nt++) {
    int col = 16*nt + l15;
    float g = ln_g[col], bb = ln_b[col], w2 = Ws2[col];
    #pragma unroll
    for (int v = 0; v < 8; v++) {
      float hn = (acc[nt][v] - mean[v]) * rs[v] * g + bb;
      hn = hn >= 0.f ? hn : NEGSL * hn;
      raw[v] += hn * w2;
    }
  }
  #pragma unroll
  for (int v = 0; v < 8; v++) {
    #pragma unroll
    for (int m = 1; m < 16; m <<= 1) raw[v] += __shfl_xor(raw[v], m);
    raw[v] += bs2[0];
  }

  // ---- GEMM2: edge_hidden = relu(ein @ Wp + bp), reuse A fragments ----
  #pragma unroll
  for (int nt = 0; nt < 8; nt++) acc[nt] = (v8f){0,0,0,0,0,0,0,0};
  const uint4* b2 = (const uint4*)(wsf + WS_WP);
  #pragma unroll
  for (int kt = 0; kt < 5; kt++) {
    #pragma unroll
    for (int nt = 0; nt < 8; nt++) {
      frag_t B; const uint4* q = b2 + ((kt*8 + nt)*64 + lane*2);
      B.u[0] = q[0]; B.u[1] = q[1];
      acc[nt] = __builtin_amdgcn_wmma_f32_16x16x32_bf16(false, A[kt].v, false, B.v,
                                                        (short)0, acc[nt], false, false);
    }
  }
  #pragma unroll
  for (int nt = 0; nt < 8; nt++) {
    int col = 16*nt + l15;
    float b = bp[col];
    #pragma unroll
    for (int v = 0; v < 8; v++) {
      float h = acc[nt][v] + b; h = h > 0.f ? h : 0.f;
      ehb[w][v + 8*half][col] = (__bf16)h;
    }
  }
  __builtin_amdgcn_wave_barrier();

  // reload edge_hidden as A fragments (K=128 -> 4 tiles)
  frag_t A2[4];
  const __bf16* rp2 = &ehb[w][l15][0];
  #pragma unroll
  for (int kt = 0; kt < 4; kt++) {
    const uint4* q = (const uint4*)(rp2 + 32*kt + 16*half);
    A2[kt].u[0] = q[0]; A2[kt].u[1] = q[1];
  }

  // ---- GEMM3 (per regime): g1 = relu(eh @ Wg1[r]); gate = sigmoid(g1 @ Wg2[r]) ----
  float rpv0 = wsf[WS_RP], rpv1 = wsf[WS_RP+1], rpv2 = wsf[WS_RP+2];
  float amp  = wsf[WS_AMP];
  float gc[8];
  #pragma unroll
  for (int v = 0; v < 8; v++) gc[v] = 0.f;
  const uint4* b3 = (const uint4*)(wsf + WS_WG1);
  for (int rr = 0; rr < 3; rr++) {
    v8f a4[4];
    #pragma unroll
    for (int nt = 0; nt < 4; nt++) a4[nt] = (v8f){0,0,0,0,0,0,0,0};
    #pragma unroll
    for (int kt = 0; kt < 4; kt++) {
      #pragma unroll
      for (int nt = 0; nt < 4; nt++) {
        frag_t B; const uint4* q = b3 + (((rr*4 + kt)*4 + nt)*64 + lane*2);
        B.u[0] = q[0]; B.u[1] = q[1];
        a4[nt] = __builtin_amdgcn_wmma_f32_16x16x32_bf16(false, A2[kt].v, false, B.v,
                                                         (short)0, a4[nt], false, false);
      }
    }
    float gp[8];
    #pragma unroll
    for (int v = 0; v < 8; v++) gp[v] = 0.f;
    #pragma unroll
    for (int nt = 0; nt < 4; nt++) {
      int col = 16*nt + l15;
      float b = bg1[rr*64 + col], w2 = Wg2[rr*64 + col];
      #pragma unroll
      for (int v = 0; v < 8; v++) {
        float g1v = a4[nt][v] + b; g1v = g1v > 0.f ? g1v : 0.f;
        gp[v] += g1v * w2;
      }
    }
    float rprob = (rr == 0) ? rpv0 : (rr == 1) ? rpv1 : rpv2;
    #pragma unroll
    for (int v = 0; v < 8; v++) {
      #pragma unroll
      for (int m = 1; m < 16; m <<= 1) gp[v] += __shfl_xor(gp[v], m);
      float gate = 1.f / (1.f + __expf(-(gp[v] + bg2[rr])));
      gc[v] += gate * rprob;
    }
  }

  // final scores (uniform per 16-lane half after butterflies)
  if (l15 == 0) {
    #pragma unroll
    for (int v = 0; v < 8; v++)
      wsf[WS_SCORES + e0 + 8*half + v] = raw[v] * gc[v] * amp;
  }
}

// ---------------- per-node softmax + top-K (one wave / node) ----------------
__global__ void topk_k(const int* __restrict__ eidx, const float* __restrict__ wsf,
                       float* __restrict__ out) {
  int gid = blockIdx.x*blockDim.x + threadIdx.x;
  int n = gid >> 5, lane = threadIdx.x & 31;
  if (n >= NODES) return;
  float s = wsf[WS_SCORES + n*DEG + lane];
  int   tg = eidx[EDGES + n*DEG + lane];
  // dedup (last-write-wins, matching scatter .set): invalid if a later edge hits same column
  bool valid = true;
  for (int j = 0; j < 32; j++) {
    int tj = __shfl(tg, j);
    if (j > lane && tj == tg) valid = false;
  }
  float sv = valid ? s : -__builtin_inff();
  float m = sv;
  #pragma unroll
  for (int k = 1; k < 32; k <<= 1) m = fmaxf(m, __shfl_xor(m, k));
  float e = valid ? __expf((s - m) * (1.f/TEMP_E)) : 0.f;
  float sum = e;
  #pragma unroll
  for (int k = 1; k < 32; k <<= 1) sum += __shfl_xor(sum, k);
  float p = valid ? e / sum : 0.f;
  // rank by (p desc, column asc) — matches lax.top_k tie-breaking
  int iv = valid ? 1 : 0, rank = 0;
  for (int j = 0; j < 32; j++) {
    float pj = __shfl(p, j); int tj = __shfl(tg, j); int vj = __shfl(iv, j);
    if (vj && (pj > p || (pj == p && tj < tg))) rank++;
  }
  int cnt = __popcll(__ballot(valid));
  if (valid) {
    out[n*KTOP + rank] = (p > W_THR) ? p : 0.f;
    out[NODES*KTOP + n*KTOP + rank] = (float)tg;
  }
  // fill remaining slots with zeros at the smallest absent columns
  int slot = cnt;
  for (int c = 0; slot < KTOP && c < 4*KTOP; c++) {
    bool pres = __any(valid && tg == c);
    if (!pres) {
      if (lane == 0) { out[n*KTOP + slot] = 0.f; out[NODES*KTOP + n*KTOP + slot] = (float)c; }
      slot++;
    }
  }
}

// ---------------- launch ----------------
extern "C" void kernel_launch(void* const* d_in, const int* in_sizes, int n_in,
                              void* d_out, int out_size, void* d_ws, size_t ws_size,
                              hipStream_t stream) {
  const float* x    = (const float*)d_in[0];
  const int*   eidx = (const int*)d_in[1];   // [2,E] (int per harness convention)
  const float* ea   = (const float*)d_in[2];
  const float* Ws1  = (const float*)d_in[3];
  const float* bs1  = (const float*)d_in[4];
  const float* ln_g = (const float*)d_in[5];
  const float* ln_b = (const float*)d_in[6];
  const float* Ws2  = (const float*)d_in[7];
  const float* bs2  = (const float*)d_in[8];
  const float* Wp   = (const float*)d_in[9];
  const float* bp   = (const float*)d_in[10];
  const float* Wr1  = (const float*)d_in[11];
  const float* br1  = (const float*)d_in[12];
  const float* Wr2  = (const float*)d_in[13];
  const float* br2  = (const float*)d_in[14];
  const float* Wg1  = (const float*)d_in[15];
  const float* bg1  = (const float*)d_in[16];
  const float* Wg2  = (const float*)d_in[17];
  const float* bg2  = (const float*)d_in[18];
  const float* Wc1  = (const float*)d_in[19];
  const float* bc1  = (const float*)d_in[20];
  const float* Wc2  = (const float*)d_in[21];
  const float* bc2  = (const float*)d_in[22];
  float* wsf = (float*)d_ws;
  float* out = (float*)d_out;

  hipLaunchKernelGGL(colmean_k,   dim3(64),   dim3(256), 0, stream, x, wsf);
  hipLaunchKernelGGL(contagion_k, dim3(1024), dim3(256), 0, stream, x, Wc1, bc1, Wc2, bc2, wsf);
  hipLaunchKernelGGL(scalars_k,   dim3(1),    dim3(128), 0, stream, Wr1, br1, Wr2, br2, wsf);
  hipLaunchKernelGGL(pack_k,      dim3(128),  dim3(256), 0, stream, Ws1, Wp, Wg1, wsf);
  hipLaunchKernelGGL(edge_k,      dim3(EDGES/16/8), dim3(256), 0, stream,
                     x, eidx, ea, bs1, ln_g, ln_b, Ws2, bs2, bp, bg1, Wg2, bg2, wsf);
  hipLaunchKernelGGL(topk_k,      dim3(NODES/8), dim3(256), 0, stream, eidx, wsf, out);
}
